// AttentionProbe_17987323035971
// MI455X (gfx1250) — compile-verified, matrix operand order checked by reading, not run on hardware
//
#include <hip/hip_runtime.h>
#include <hip/hip_bf16.h>

// ---------------------------------------------------------------------------
// Sliding-window attention probe, MI455X (gfx1250), wave32 + WMMA bf16 + TDM.
// B=2, S=2048, D=2048, H=16, DP=128, WINDOW=512
// ---------------------------------------------------------------------------

#define BT      2
#define S_LEN   2048
#define D_MODEL 2048
#define NH      16
#define DPH     128
#define WIN     512

typedef __attribute__((ext_vector_type(16))) __bf16 bf16x16;
typedef __attribute__((ext_vector_type(8)))  __bf16 bf16x8;
typedef __attribute__((ext_vector_type(8)))  float  f32x8;
typedef __attribute__((ext_vector_type(4)))  unsigned u32x4;
typedef __attribute__((ext_vector_type(8)))  int     i32x8;
typedef __attribute__((ext_vector_type(4)))  int     i32x4;

#if __has_builtin(__builtin_amdgcn_tensor_load_to_lds) && \
    __has_builtin(__builtin_amdgcn_s_wait_tensorcnt)
#define HAVE_TDM 1
#else
#define HAVE_TDM 0
#endif

static __device__ __forceinline__ __bf16 f2bf(float f) {
  unsigned u = __builtin_bit_cast(unsigned, f);
  unsigned r = (u + 0x7FFFu + ((u >> 16) & 1u)) >> 16;
  return __builtin_bit_cast(__bf16, (unsigned short)r);
}

// Load a 16x32 bf16 WMMA fragment (A and B layouts are lane-symmetric):
//   lanes 0..15 : row = rowBase+lane,    cols k0+0..7   and k0+16..23
//   lanes 16..31: row = rowBase+lane-16, cols k0+8..15  and k0+24..31
static __device__ __forceinline__ bf16x16
load_frag16x32(const __bf16* base, int stride, int rowBase, int k0) {
  const int lane = threadIdx.x & 31;
  const __bf16* p = base + (size_t)(rowBase + (lane & 15)) * stride + k0 + ((lane & 16) ? 8 : 0);
  bf16x8 lo = *(const bf16x8*)p;
  bf16x8 hi = *(const bf16x8*)(p + 16);
  bf16x16 r;
#pragma unroll
  for (int i = 0; i < 8; ++i) { r[i] = lo[i]; r[8 + i] = hi[i]; }
  return r;
}

#if HAVE_TDM
// Issue a TDM 2D tile load: bf16 elements, tile_d1 rows of tile_d0 elements,
// row stride d0_stride (elements). HW padding: +16B after every 256B row, so a
// 128-wide bf16 row lands at LDS stride 136 bf16 (bank-conflict-free frags).
static __device__ __forceinline__ void
tdm_load_2d_bf16(const void* gaddr, unsigned lds_off,
                 unsigned tensor_d0, unsigned tensor_d1, unsigned d0_stride,
                 unsigned tile_d0, unsigned tile_d1) {
  unsigned long long ga = (unsigned long long)(size_t)gaddr;
  u32x4 g0;
  g0[0] = 1u;                                            // count=1, user mode
  g0[1] = lds_off;                                       // LDS byte address
  g0[2] = (unsigned)(ga & 0xFFFFFFFFu);                  // global addr lo
  g0[3] = (unsigned)((ga >> 32) & 0x01FFFFFFu) | (2u << 30);  // addr hi | type=2
  i32x8 g1;
  g1[0] = (int)((1u << 16)      // data_size = 1 (2 bytes)
              | (1u << 20)      // pad_enable
              | (5u << 22)      // pad_interval: 2^(5+1)=64 DW = 256B
              | (3u << 25));    // pad_amount: 4 DW = 16B
  g1[1] = (int)((tensor_d0 & 0xFFFFu) << 16);                    // dim0 lo16
  g1[2] = (int)(((tensor_d0 >> 16) & 0xFFFFu) | ((tensor_d1 & 0xFFFFu) << 16));
  g1[3] = (int)(((tensor_d1 >> 16) & 0xFFFFu) | ((tile_d0 & 0xFFFFu) << 16));
  g1[4] = (int)(tile_d1 & 0xFFFFu);                              // tile_dim2=0
  g1[5] = (int)d0_stride;                                        // stride lo32
  g1[6] = 0;                                                     // stride hi, dim1_stride=0
  g1[7] = 0;
  i32x4 z4 = {0, 0, 0, 0};
  i32x8 z8 = {0, 0, 0, 0, 0, 0, 0, 0};
  __builtin_amdgcn_tensor_load_to_lds(g0, g1, z4, z4, z8, 0);
}
#endif

// ---------------------------------------------------------------------------
// fp32 -> bf16 conversion (bandwidth-bound prepass)
// ---------------------------------------------------------------------------
__global__ void cvt_f32_bf16(const float* __restrict__ src, __bf16* __restrict__ dst, int n) {
  int i = blockIdx.x * blockDim.x + threadIdx.x;
  int stride = gridDim.x * blockDim.x;
  for (; i < n; i += stride) dst[i] = f2bf(src[i]);
}

// ---------------------------------------------------------------------------
// Output init: out[b,s] = bo
// ---------------------------------------------------------------------------
__global__ void init_out(float* __restrict__ out, const float* __restrict__ bo, int n) {
  int i = blockIdx.x * blockDim.x + threadIdx.x;
  if (i < n) out[i] = bo[0];
}

// ---------------------------------------------------------------------------
// QKV projection GEMM:  C(M=4096, N=2048) = xb(M,K=2048) * wb(K,N) + bias
// blockIdx.z selects {q,k,v}. 128x128 tile per block, 8 waves, each wave
// computes a 32x64 region = 2x4 WMMA 16x16 tiles, K stepped by 32.
// q,k stored [b][h][s][dp] bf16; v stored transposed [b][h][dp][s] bf16.
// ---------------------------------------------------------------------------
__global__ __launch_bounds__(256)
void qkv_gemm(const __bf16* __restrict__ xb,
              const __bf16* __restrict__ wqb, const __bf16* __restrict__ wkb,
              const __bf16* __restrict__ wvb,
              const float* __restrict__ bq, const float* __restrict__ bk,
              const float* __restrict__ bv,
              __bf16* __restrict__ qo, __bf16* __restrict__ ko, __bf16* __restrict__ vo) {
  const int z = blockIdx.z;
  const __bf16* wb  = (z == 0) ? wqb : (z == 1) ? wkb : wvb;
  const float* bias = (z == 0) ? bq  : (z == 1) ? bk  : bv;
  __bf16* dst       = (z == 0) ? qo  : (z == 1) ? ko  : vo;

  const int n0 = blockIdx.x * 128;
  const int m0 = blockIdx.y * 128;
  const int t  = threadIdx.x;
  const int lane = t & 31;
  const int wave = t >> 5;
  const int wm = wave & 3;   // 4 m-subtiles of 32 rows
  const int wn = wave >> 2;  // 2 n-subtiles of 64 cols

  __shared__ __bf16 Al[128 * 40];   // A tile 128x32, padded stride 40
  __shared__ __bf16 Btl[128 * 40];  // B tile transposed: [n][k], stride 40

  f32x8 acc[2][4];
#pragma unroll
  for (int a = 0; a < 2; ++a)
#pragma unroll
    for (int b = 0; b < 4; ++b)
#pragma unroll
      for (int j = 0; j < 8; ++j) acc[a][b][j] = 0.0f;

  for (int k0 = 0; k0 < D_MODEL; k0 += 32) {
    __syncthreads();
    {  // A tile: thread -> row t>>1, 16 cols at (t&1)*16
      int r = t >> 1, c = (t & 1) * 16;
      const __bf16* g = xb + (size_t)(m0 + r) * D_MODEL + k0 + c;
      bf16x8 v0 = *(const bf16x8*)g;
      bf16x8 v1 = *(const bf16x8*)(g + 8);
      *(bf16x8*)&Al[r * 40 + c]     = v0;
      *(bf16x8*)&Al[r * 40 + c + 8] = v1;
    }
    {  // B tile, transposed into LDS: thread -> k = t>>3, n chunk (t&7)*16
      int kk = t >> 3, nc = (t & 7) * 16;
      const __bf16* g = wb + (size_t)(k0 + kk) * D_MODEL + n0 + nc;
      bf16x8 v0 = *(const bf16x8*)g;
      bf16x8 v1 = *(const bf16x8*)(g + 8);
#pragma unroll
      for (int e = 0; e < 8; ++e) {
        Btl[(nc + e) * 40 + kk]     = v0[e];
        Btl[(nc + 8 + e) * 40 + kk] = v1[e];
      }
    }
    // Prefetch next K-step tiles into cache (global_prefetch_b8).
    if (k0 + 32 < D_MODEL) {
      __builtin_prefetch(xb + (size_t)(m0 + (t >> 1)) * D_MODEL + k0 + 32 + (t & 1) * 16, 0, 3);
      __builtin_prefetch(wb + (size_t)(k0 + 32 + (t >> 3)) * D_MODEL + n0 + (t & 7) * 16, 0, 3);
    }
    __syncthreads();

    bf16x16 af[2], bf[4];
#pragma unroll
    for (int tm = 0; tm < 2; ++tm) af[tm] = load_frag16x32(Al, 40, wm * 32 + tm * 16, 0);
#pragma unroll
    for (int tn = 0; tn < 4; ++tn) bf[tn] = load_frag16x32(Btl, 40, wn * 64 + tn * 16, 0);
#pragma unroll
    for (int tm = 0; tm < 2; ++tm)
#pragma unroll
      for (int tn = 0; tn < 4; ++tn)
        acc[tm][tn] = __builtin_amdgcn_wmma_f32_16x16x32_bf16(
            false, af[tm], false, bf[tn], (short)0, acc[tm][tn], false, false);
  }

  // Epilogue: bias add, bf16 convert, scatter to head-major layouts.
  const int hi = (lane & 16) ? 8 : 0;
#pragma unroll
  for (int tm = 0; tm < 2; ++tm)
#pragma unroll
    for (int tn = 0; tn < 4; ++tn)
#pragma unroll
      for (int j = 0; j < 8; ++j) {
        int row = m0 + wm * 32 + tm * 16 + j + hi;      // (b,s)
        int col = n0 + wn * 64 + tn * 16 + (lane & 15); // (h,dp)
        float v = acc[tm][tn][j] + bias[col];
        int bb = row >> 11, s = row & (S_LEN - 1);
        int h  = col >> 7,  dp = col & (DPH - 1);
        size_t idx = (z == 2)
            ? (((size_t)(bb * NH + h) * DPH + dp) * S_LEN + s)   // V transposed
            : (((size_t)(bb * NH + h) * S_LEN + s) * DPH + dp);  // Q,K row-major
        dst[idx] = f2bf(v);
      }
}

// ---------------------------------------------------------------------------
// Fused sliding-window flash attention + output projection.
// Grid: (S/128, H, B); block 256 = 8 waves; wave owns 16 q rows.
// Computes S^T = K * Q^T so that q lives in lanes (per-lane online softmax)
// and the P fragments for P*V are built with pure per-lane packing.
// K/V tiles are DMA'd into LDS by the Tensor Data Mover (one wave issues,
// TENSORcnt-pipelined: V(kt) overlaps score WMMAs, K(kt+1) overlaps P*V).
// Epilogue folds `o @ wo` and atomicAdds into out[b][s].
// ---------------------------------------------------------------------------
__global__ __launch_bounds__(256)
void attn_swin(const __bf16* __restrict__ qb, const __bf16* __restrict__ kb,
               const __bf16* __restrict__ vtb, const float* __restrict__ wo,
               float* __restrict__ out) {
  const int b = blockIdx.z, h = blockIdx.y, qt = blockIdx.x;
  const int q0 = qt * 128;
  const int t = threadIdx.x, wave = t >> 5, lane = t & 31;
  const int hi = (lane & 16) ? 8 : 0;

  const __bf16* Q  = qb  + (size_t)(b * NH + h) * S_LEN * DPH;  // [s][dp]
  const __bf16* K  = kb  + (size_t)(b * NH + h) * S_LEN * DPH;  // [s][dp]
  const __bf16* Vt = vtb + (size_t)(b * NH + h) * DPH * S_LEN;  // [dp][s]

  __shared__ __bf16 bufK[128 * 136];  // K tile   (34 KB, stride 136 via TDM pad)
  __shared__ __bf16 bufV[128 * 136];  // V^T tile (34 KB)

  // Q B-fragments for this wave's 16 q rows (resident whole kernel).
  bf16x16 qf[4];
#pragma unroll
  for (int kk = 0; kk < 4; ++kk)
    qf[kk] = load_frag16x32(Q, DPH, q0 + wave * 16, kk * 32);

  f32x8 O[8];
#pragma unroll
  for (int n = 0; n < 8; ++n)
#pragma unroll
    for (int j = 0; j < 8; ++j) O[n][j] = 0.0f;

  float m = -1e30f, l = 0.0f;
  const int qrow = q0 + wave * 16 + (lane & 15);  // this lane's q index

  float wol[8];
#pragma unroll
  for (int n = 0; n < 8; ++n) wol[n] = wo[h * DPH + n * 16 + (lane & 15)];

  int ktlo = qt - 4; if (ktlo < 0) ktlo = 0;

#if HAVE_TDM
  const unsigned ldsK = (unsigned)(size_t)&bufK[0];
  const unsigned ldsV = (unsigned)(size_t)&bufV[0];
  if (t < 32)  // preload first K tile
    tdm_load_2d_bf16(K + (size_t)(ktlo * 128) * DPH, ldsK, DPH, S_LEN, DPH, 128, 128);
#endif

  for (int kt = ktlo; kt <= qt; ++kt) {
    const int kv0 = kt * 128;

    __syncthreads();  // (A) previous iteration's reads of bufK/bufV complete
#if HAVE_TDM
    if (t < 32) {
      tdm_load_2d_bf16(Vt + kv0, ldsV, S_LEN, DPH, S_LEN, 128, 128);  // overlap scores
      __builtin_amdgcn_s_wait_tensorcnt(1);                           // K(kt) done
    }
#else
    {  // manual fill: K tile rows + V^T tile rows
      int r = t >> 1, c = (t & 1) * 64;
      const __bf16* gk = K + (size_t)(kv0 + r) * DPH + c;
      const __bf16* gv = Vt + (size_t)r * S_LEN + kv0 + c;
#pragma unroll
      for (int j = 0; j < 8; ++j) {
        *(bf16x8*)&bufK[r * 136 + c + j * 8] = *(const bf16x8*)(gk + j * 8);
        *(bf16x8*)&bufV[r * 136 + c + j * 8] = *(const bf16x8*)(gv + j * 8);
      }
    }
#endif
    __syncthreads();  // (B) K tile visible

    // Scores^T: 8 kv-subtiles of 16, each = K_sub(16x32dp) x Q^T, 4 dp-steps.
    f32x8 sc[8];
#pragma unroll
    for (int i = 0; i < 8; ++i) {
      f32x8 c;
#pragma unroll
      for (int j = 0; j < 8; ++j) c[j] = 0.0f;
#pragma unroll
      for (int kk = 0; kk < 4; ++kk) {
        bf16x16 kf = load_frag16x32(bufK, 136, i * 16, kk * 32);
        c = __builtin_amdgcn_wmma_f32_16x16x32_bf16(false, kf, false, qf[kk],
                                                    (short)0, c, false, false);
      }
      sc[i] = c;
    }

    // Mask + scale + tile max (q in lanes; kv in (reg, half)).
    const float SCALE = 0.08838834764831845f;  // 1/sqrt(128)
    float mt = -1e30f;
#pragma unroll
    for (int i = 0; i < 8; ++i)
#pragma unroll
      for (int j = 0; j < 8; ++j) {
        int kv = kv0 + i * 16 + j + hi;
        float v = sc[i][j] * SCALE;
        bool ok = (qrow >= kv) && (qrow - kv < WIN);
        v = ok ? v : -1e30f;
        sc[i][j] = v;
        mt = fmaxf(mt, v);
      }
    mt = fmaxf(mt, __shfl_xor(mt, 16, 32));
    float mn = fmaxf(m, mt);
    float alpha = __expf(m - mn);
    m = mn;

    float lt = 0.0f;
#pragma unroll
    for (int i = 0; i < 8; ++i)
#pragma unroll
      for (int j = 0; j < 8; ++j) {
        float v = sc[i][j];
        float p = (v > -1e29f) ? __expf(v - mn) : 0.0f;
        sc[i][j] = p;
        lt += p;
      }
    lt += __shfl_xor(lt, 16, 32);
    l = l * alpha + lt;

    // Build P A-fragments: pure per-lane packing from S^T C-layout.
    bf16x16 pf[4];
#pragma unroll
    for (int kk = 0; kk < 4; ++kk)
#pragma unroll
      for (int j = 0; j < 8; ++j) {
        pf[kk][j]     = f2bf(sc[2 * kk][j]);
        pf[kk][8 + j] = f2bf(sc[2 * kk + 1][j]);
      }

    // Rescale O accumulators by alpha[q]; q = reg + (half ? 8 : 0).
#pragma unroll
    for (int j = 0; j < 8; ++j) {
      float aj = __shfl(alpha, j + hi, 32);
#pragma unroll
      for (int n = 0; n < 8; ++n) O[n][j] *= aj;
    }

    __syncthreads();  // (C) all waves done reading bufK
#if HAVE_TDM
    if (t < 32) {
      if (kt < qt) {
        tdm_load_2d_bf16(K + (size_t)((kt + 1) * 128) * DPH, ldsK,
                         DPH, S_LEN, DPH, 128, 128);  // overlaps P*V below
        __builtin_amdgcn_s_wait_tensorcnt(1);         // V(kt) done
      } else {
        __builtin_amdgcn_s_wait_tensorcnt(0);         // V(kt) done (last iter)
      }
    }
#endif
    __syncthreads();  // (D) V tile visible

    // O += P * V : 8 dp-subtiles x 4 kv-steps.
#pragma unroll
    for (int n = 0; n < 8; ++n) {
      f32x8 c = O[n];
#pragma unroll
      for (int kk = 0; kk < 4; ++kk) {
        bf16x16 vf = load_frag16x32(bufV, 136, n * 16, kk * 32);
        c = __builtin_amdgcn_wmma_f32_16x16x32_bf16(false, pf[kk], false, vf,
                                                    (short)0, c, false, false);
      }
      O[n] = c;
    }
  }

  // Finalize: normalize, fold wo, reduce over dp, atomicAdd into out[b][s].
  float inv = 1.0f / l;
  float accj[8];
#pragma unroll
  for (int j = 0; j < 8; ++j) {
    float s = 0.0f;
#pragma unroll
    for (int n = 0; n < 8; ++n) s += O[n][j] * wol[n];
    s *= __shfl(inv, j + hi, 32);
    s += __shfl_xor(s, 8, 32);
    s += __shfl_xor(s, 4, 32);
    s += __shfl_xor(s, 2, 32);
    s += __shfl_xor(s, 1, 32);
    accj[j] = s;
  }
  if ((lane & 15) == 0) {
#pragma unroll
    for (int j = 0; j < 8; ++j)
      atomicAdd(&out[(size_t)b * S_LEN + q0 + wave * 16 + j + hi], accj[j]);
  }
}

// ---------------------------------------------------------------------------
extern "C" void kernel_launch(void* const* d_in, const int* in_sizes, int n_in,
                              void* d_out, int out_size, void* d_ws, size_t ws_size,
                              hipStream_t stream) {
  (void)in_sizes; (void)n_in; (void)out_size; (void)ws_size;
  const float* x  = (const float*)d_in[0];
  const float* wq = (const float*)d_in[1];
  const float* bq = (const float*)d_in[2];
  const float* wk = (const float*)d_in[3];
  const float* bk = (const float*)d_in[4];
  const float* wv = (const float*)d_in[5];
  const float* bv = (const float*)d_in[6];
  const float* wo = (const float*)d_in[7];
  const float* bo = (const float*)d_in[8];
  float* out = (float*)d_out;

  const size_t NX = (size_t)BT * S_LEN * D_MODEL;       // 8,388,608
  const size_t NW = (size_t)D_MODEL * NH * DPH;         // 4,194,304
  const size_t NQ = (size_t)BT * NH * S_LEN * DPH;      // 8,388,608

  char* ws = (char*)d_ws;
  size_t off = 0;
  __bf16* xb  = (__bf16*)(ws + off); off += NX * 2;
  __bf16* wqb = (__bf16*)(ws + off); off += NW * 2;
  __bf16* wkb = (__bf16*)(ws + off); off += NW * 2;
  __bf16* wvb = (__bf16*)(ws + off); off += NW * 2;
  __bf16* qo  = (__bf16*)(ws + off); off += NQ * 2;
  __bf16* ko  = (__bf16*)(ws + off); off += NQ * 2;
  __bf16* vo  = (__bf16*)(ws + off); off += NQ * 2;   // ~92 MB total

  cvt_f32_bf16<<<4096, 256, 0, stream>>>(x,  xb,  (int)NX);
  cvt_f32_bf16<<<2048, 256, 0, stream>>>(wq, wqb, (int)NW);
  cvt_f32_bf16<<<2048, 256, 0, stream>>>(wk, wkb, (int)NW);
  cvt_f32_bf16<<<2048, 256, 0, stream>>>(wv, wvb, (int)NW);

  qkv_gemm<<<dim3(16, 32, 3), 256, 0, stream>>>(xb, wqb, wkb, wvb,
                                                bq, bk, bv, qo, ko, vo);

  init_out<<<16, 256, 0, stream>>>(out, bo, BT * S_LEN);

  attn_swin<<<dim3(S_LEN / 128, NH, BT), 256, 0, stream>>>(qo, ko, vo, wo, out);
}
